// DeepJ_8727373545747
// MI455X (gfx1250) — compile-verified, matrix operand order checked by reading, not run on hardware
//
#include <hip/hip_runtime.h>
#include <cstdint>

// ---------------- types ----------------
typedef __bf16 bf16;
typedef __attribute__((ext_vector_type(16))) __bf16 v16bf;
typedef __attribute__((ext_vector_type(8)))  float  v8f;

#define B_SZ 1024
#define NN   48
#define ROWS (B_SZ*NN)      // 49152
#define H_TA 256
#define U_NA 128

__device__ __forceinline__ float sigm(float x){ return 1.0f/(1.0f + __expf(-x)); }

__device__ __forceinline__ v8f vzero8(){
  v8f z;
  #pragma unroll
  for (int i = 0; i < 8; ++i) z[i] = 0.0f;
  return z;
}

// A fragment: 16x32 bf16 tile (M x K), row-major with stride lda (LDS-resident).
// ISA 7.12.2: lanes 0-15 & 16-31 hold M=lane&15; lane half selects K offset +8;
// VGPR v packs K pairs (v<4 -> K base 0, v>=4 -> K base 16).
// Pairs are contiguous in K -> lowers to 2x ds_load_b128 per lane.
__device__ __forceinline__ v16bf load_a_frag(const bf16* A, int lda){
  const int lane = threadIdx.x & 31;
  const int m  = lane & 15;
  const int kh = (lane >> 4) << 3;
  v16bf r;
  #pragma unroll
  for (int v = 0; v < 8; ++v){
    int k = ((v & 4) << 2) + kh + ((v & 3) << 1);
    r[2*v]   = A[m*lda + k];
    r[2*v+1] = A[m*lda + k + 1];
  }
  return r;
}

// B fragment from PRE-SWIZZLED weights: each 32x16 (KxN) tile is stored
// fragment-major: lane L's 16 bf16 values contiguous at blk + L*16.
// -> one aligned 32-byte vector load per lane (2x global_load_b128).
__device__ __forceinline__ v16bf load_b_swz(const bf16* blk){
  const int lane = threadIdx.x & 31;
  return *(const v16bf*)(blk + lane*16);
}

__device__ __forceinline__ v8f wmma_bf16(v16bf a, v16bf b, v8f c){
  return __builtin_amdgcn_wmma_f32_16x16x32_bf16(false, a, false, b, (short)0, c, false, false);
}

// ---------------- weight packing into WMMA-fragment-major layout ----------------
// Source W is [N][Kin] row-major (torch weight [out,in]); logical B = W^T [K][N].
// Output: tiles (nt, kt) of 32x16, tile index = nt*nKT + kt, 512 bf16 per tile,
// element (lane, j): j=2v+odd -> B[k][n] with
//   n  = nt*16 + (lane&15)
//   kh = (lane>>4)*8
//   k  = kt*32 + ((v&4)<<2) + kh + ((v&3)<<1) + odd
__global__ void pack_w_swz(const float* __restrict__ W, bf16* __restrict__ out,
                           int N, int Kin, int Kpad){
  int idx = blockIdx.x*blockDim.x + threadIdx.x;
  int total = N*Kpad;
  if (idx >= total) return;
  int nKT    = Kpad >> 5;
  int blk    = idx >> 9;           // /512
  int within = idx & 511;
  int lane   = within >> 4;
  int j      = within & 15;
  int nt = blk / nKT;
  int kt = blk - nt*nKT;
  int n  = nt*16 + (lane & 15);
  int kh = (lane >> 4) << 3;
  int v  = j >> 1;
  int k  = kt*32 + ((v & 4) << 2) + kh + ((v & 3) << 1) + (j & 1);
  out[idx] = (k < Kin) ? (bf16)W[n*Kin + k] : (bf16)0.0f;
}

__global__ void combine_bias(const float* __restrict__ a, const float* __restrict__ b,
                             float* __restrict__ out, int n){
  int i = blockIdx.x*blockDim.x + threadIdx.x;
  if (i < n) out[i] = a[i] + b[i];
}

// tail[j][n] = na_Wih0[n][256+j]  (the 3 recurrent-input columns)
__global__ void pack_tail(const float* __restrict__ W, float* __restrict__ tail){
  int i = blockIdx.x*blockDim.x + threadIdx.x;
  if (i < 3*512){
    int j = i / 512, n = i % 512;
    tail[i] = W[n*259 + 256 + j];
  }
}

// ---------------- feature assembly ----------------
// feat[b*48+note][0..95] = [pitch_pos | pitch_class(12) | beat(16) | tanh(conv)(32) | chord(12) | 0pad(23)]
__global__ void build_features(const float* __restrict__ note_input,
                               const float* __restrict__ beat_in,
                               const float* __restrict__ beat_W,
                               const float* __restrict__ beat_b,
                               const float* __restrict__ vic_W,
                               const float* __restrict__ vic_b,
                               bf16* __restrict__ feat){
  long idx = (long)blockIdx.x*blockDim.x + threadIdx.x;
  const long total = (long)ROWS*96;
  if (idx >= total) return;
  int  c    = (int)(idx % 96);
  long row  = idx / 96;
  int  note = (int)(row % NN);
  int  b    = (int)(row / NN);
  float val = 0.0f;
  if (c == 0){
    val = (float)note / 48.0f;
  } else if (c < 13){
    val = ((note % 12) == (c-1)) ? 1.0f : 0.0f;
  } else if (c < 29){
    int u = c - 13;
    float s = beat_b[u];
    #pragma unroll
    for (int k = 0; k < 16; ++k) s += beat_in[b*16+k]*beat_W[u*16+k];
    val = s;
  } else if (c < 61){
    int u = c - 29;
    float s = vic_b[u];
    for (int ch = 0; ch < 3; ++ch)
      #pragma unroll 5
      for (int t = 0; t < 25; ++t){
        int p = note + t - 12;
        if (p >= 0 && p < NN)
          s += note_input[(b*NN + p)*3 + ch] * vic_W[(u*3 + ch)*25 + t];
      }
    val = tanhf(s);
  } else if (c < 73){
    int j = c - 61;
    float s = 0.0f;
    #pragma unroll
    for (int o = 0; o < 4; ++o) s += note_input[(b*NN + j*4 + o)*3 + 0];
    val = s * 0.25f;
  }
  feat[row*96 + c] = (bf16)val;
}

// ---------------- TimeAxis: zero-state LSTM layer = GEMM (i/g/o only) + pointwise ----------------
// X [M][KPAD] bf16, Wswz swizzled [64 ntiles][nKT ktiles][512], bias [1024] f32, H [M][256] bf16
// Block = one 16-row M tile, 16 waves; wave w owns columns w*16..w*16+15 of each live gate.
template<int KPAD>
__global__ void __launch_bounds__(512)
lstm_zero_gemm(const bf16* __restrict__ X, const bf16* __restrict__ Wswz,
               const float* __restrict__ bias, bf16* __restrict__ H){
  constexpr int nKT = KPAD/32;
  __shared__ __align__(16) bf16 As[16*KPAD];
  const int mbase = blockIdx.x * 16;
  { // vectorized cooperative stage of the contiguous 16-row A tile
    const uint4* src = (const uint4*)(X + (size_t)mbase*KPAD);
    uint4* dst = (uint4*)As;
    for (int i = threadIdx.x; i < 2*KPAD; i += blockDim.x) dst[i] = src[i];
  }
  __syncthreads();

  const int wave = threadIdx.x >> 5;      // 0..15
  const int lane = threadIdx.x & 31;
  // n-tile indices within the 1024-wide (64-tile) gate matrix; f-gate dead (c_prev==0)
  const bf16* Bi = Wswz + (size_t)(( 0 + wave)*nKT)*512;   // i gate
  const bf16* Bg = Wswz + (size_t)((32 + wave)*nKT)*512;   // g gate
  const bf16* Bo = Wswz + (size_t)((48 + wave)*nKT)*512;   // o gate
  v8f acc_i = vzero8(), acc_g = vzero8(), acc_o = vzero8();

  #pragma unroll
  for (int kt = 0; kt < nKT; ++kt){
    v16bf a = load_a_frag(&As[kt*32], KPAD);
    if (kt + 1 < nKT) __builtin_prefetch(Bi + (kt+1)*512, 0, 1);
    acc_i = wmma_bf16(a, load_b_swz(Bi + kt*512), acc_i);
    acc_g = wmma_bf16(a, load_b_swz(Bg + kt*512), acc_g);
    acc_o = wmma_bf16(a, load_b_swz(Bo + kt*512), acc_o);
  }
  const int n  = lane & 15;
  const int rb = (lane >> 4) << 3;   // C layout: VGPR e -> M = rb + e
  const int ci = 0*H_TA + wave*16, cg = 2*H_TA + wave*16, co = 3*H_TA + wave*16;
  const float b_i = bias[ci + n], b_g = bias[cg + n], b_o = bias[co + n];
  #pragma unroll
  for (int e = 0; e < 8; ++e){
    long m = mbase + rb + e;
    float gi = sigm(acc_i[e] + b_i);
    float gg = tanhf(acc_g[e] + b_g);
    float go = sigm(acc_o[e] + b_o);
    H[m*H_TA + wave*16 + n] = (bf16)(go * tanhf(gi * gg));
  }
}

// ---------------- NoteAxis input projection: Y = nf @ Wx  (f32 pre-activations) ----------------
__global__ void __launch_bounds__(512)
gemm_xproj(const bf16* __restrict__ X, const bf16* __restrict__ Wswz,
           float* __restrict__ Y){
  constexpr int nKT = 8;   // K=256
  __shared__ __align__(16) bf16 As[16*256];
  const int mbase = blockIdx.x * 16;
  {
    const uint4* src = (const uint4*)(X + (size_t)mbase*256);
    uint4* dst = (uint4*)As;
    for (int i = threadIdx.x; i < 512; i += blockDim.x) dst[i] = src[i];
  }
  __syncthreads();
  const int wave = threadIdx.x >> 5, lane = threadIdx.x & 31;
  const bf16* B0 = Wswz + (size_t)(( wave      )*nKT)*512;
  const bf16* B1 = Wswz + (size_t)(((wave + 16))*nKT)*512;
  v8f a0 = vzero8(), a1 = vzero8();
  #pragma unroll
  for (int kt = 0; kt < nKT; ++kt){
    v16bf a = load_a_frag(&As[kt*32], 256);
    if (kt + 1 < nKT) __builtin_prefetch(B0 + (kt+1)*512, 0, 1);
    a0 = wmma_bf16(a, load_b_swz(B0 + kt*512), a0);
    a1 = wmma_bf16(a, load_b_swz(B1 + kt*512), a1);
  }
  const int n = lane & 15, rb = (lane >> 4) << 3;
  const int c0 = wave*16, c1 = (wave+16)*16;
  #pragma unroll
  for (int e = 0; e < 8; ++e){
    long m = mbase + rb + e;
    Y[m*512 + c0 + n] = a0[e];
    Y[m*512 + c1 + n] = a1[e];
  }
}

// ---------------- NoteAxis persistent recurrence ----------------
// 32 blocks x 512 threads; block owns 32 batch rows for all 48 steps.
// h0/h1 in LDS (bf16), c0/c1 in per-wave registers. Wave = (m tile 0..1, u group 0..7).
// Swizzled recurrent weights: N=512 -> 32 ntiles, K=128 -> nKT=4; gate g tile = g*8+u.
__global__ void __launch_bounds__(512)
note_axis_kernel(const float* __restrict__ Xp,     // [ROWS][512] pre-activations
                 const bf16*  __restrict__ Whh0,   // swizzled [32][4][512]
                 const bf16*  __restrict__ Wih1,   // swizzled
                 const bf16*  __restrict__ Whh1,   // swizzled
                 const float* __restrict__ bias0,  // [512]
                 const float* __restrict__ bias1,  // [512]
                 const float* __restrict__ tail,   // [3][512]
                 const float* __restrict__ cond,   // [B][48][3]
                 const float* __restrict__ outW,   // [3][128]
                 const float* __restrict__ outB,   // [3]
                 float* __restrict__ out)          // [B][48][3]
{
  __shared__ __align__(16) bf16  h0s[32*128];
  __shared__ __align__(16) bf16  h1s[32*128];
  __shared__ float s3[32*3];

  const int wave = threadIdx.x >> 5;
  const int lane = threadIdx.x & 31;
  const int m    = wave >> 3;            // 0..1
  const int u    = wave & 7;             // 0..7
  const int n    = lane & 15;
  const int rb   = (lane >> 4) << 3;
  const int rloc0 = m*16 + rb;           // local row base (+e)
  const int bbase = blockIdx.x * 32;

  for (int i = threadIdx.x; i < 32*128; i += blockDim.x){
    h0s[i] = (bf16)0.0f; h1s[i] = (bf16)0.0f;
  }
  v8f c0 = vzero8(), c1 = vzero8();

  const int cI = 0*U_NA + u*16;
  const int cF = 1*U_NA + u*16;
  const int cG = 2*U_NA + u*16;
  const int cO = 3*U_NA + u*16;
  // swizzled tile bases: gate g, group u -> ntile g*8+u, nKT=4 ktiles of 512
  const size_t tI = (size_t)(( 0 + u)*4)*512;
  const size_t tF = (size_t)(( 8 + u)*4)*512;
  const size_t tG = (size_t)((16 + u)*4)*512;
  const size_t tO = (size_t)((24 + u)*4)*512;

  const float b0i = bias0[cI+n], b0f = bias0[cF+n], b0g = bias0[cG+n], b0o = bias0[cO+n];
  const float b1i = bias1[cI+n], b1f = bias1[cF+n], b1g = bias1[cG+n], b1o = bias1[cO+n];
  float tlI[3], tlF[3], tlG[3], tlO[3];
  #pragma unroll
  for (int j = 0; j < 3; ++j){
    tlI[j] = tail[j*512 + cI + n];
    tlF[j] = tail[j*512 + cF + n];
    tlG[j] = tail[j*512 + cG + n];
    tlO[j] = tail[j*512 + cO + n];
  }
  __syncthreads();

  for (int step = 0; step < NN; ++step){
    // stage shifted condition notes for this step
    if (threadIdx.x < 96){
      int r = threadIdx.x / 3, j = threadIdx.x % 3;
      s3[threadIdx.x] = (step == 0) ? 0.0f
                                    : cond[((bbase + r)*NN + (step-1))*3 + j];
    }
    __syncthreads();

    // ---- layer 0: gates = Xproj + bias + tail*shifted + h0_prev @ Whh0 ----
    v8f gi, gf, gg, go;
    #pragma unroll
    for (int e = 0; e < 8; ++e){
      int rl = rloc0 + e;
      float s0 = s3[rl*3+0], s1 = s3[rl*3+1], s2 = s3[rl*3+2];
      int grow = ((bbase + rl)*NN + step)*512;
      gi[e] = Xp[grow + cI + n] + b0i + s0*tlI[0] + s1*tlI[1] + s2*tlI[2];
      gf[e] = Xp[grow + cF + n] + b0f + s0*tlF[0] + s1*tlF[1] + s2*tlF[2];
      gg[e] = Xp[grow + cG + n] + b0g + s0*tlG[0] + s1*tlG[1] + s2*tlG[2];
      go[e] = Xp[grow + cO + n] + b0o + s0*tlO[0] + s1*tlO[1] + s2*tlO[2];
    }
    #pragma unroll
    for (int kt = 0; kt < 4; ++kt){
      v16bf a = load_a_frag(&h0s[(m*16)*128 + kt*32], 128);
      gi = wmma_bf16(a, load_b_swz(Whh0 + tI + kt*512), gi);
      gf = wmma_bf16(a, load_b_swz(Whh0 + tF + kt*512), gf);
      gg = wmma_bf16(a, load_b_swz(Whh0 + tG + kt*512), gg);
      go = wmma_bf16(a, load_b_swz(Whh0 + tO + kt*512), go);
    }
    float h0v[8];
    #pragma unroll
    for (int e = 0; e < 8; ++e){
      float cc = sigm(gf[e])*c0[e] + sigm(gi[e])*tanhf(gg[e]);
      c0[e] = cc;
      h0v[e] = sigm(go[e])*tanhf(cc);
    }
    __syncthreads();                      // all h0_prev reads done
    #pragma unroll
    for (int e = 0; e < 8; ++e)
      h0s[(rloc0+e)*128 + u*16 + n] = (bf16)h0v[e];
    __syncthreads();

    // ---- layer 1: gates = h0_new @ Wih1 + h1_prev @ Whh1 + bias ----
    #pragma unroll
    for (int e = 0; e < 8; ++e){ gi[e]=b1i; gf[e]=b1f; gg[e]=b1g; go[e]=b1o; }
    #pragma unroll
    for (int kt = 0; kt < 4; ++kt){
      v16bf a0 = load_a_frag(&h0s[(m*16)*128 + kt*32], 128);
      gi = wmma_bf16(a0, load_b_swz(Wih1 + tI + kt*512), gi);
      gf = wmma_bf16(a0, load_b_swz(Wih1 + tF + kt*512), gf);
      gg = wmma_bf16(a0, load_b_swz(Wih1 + tG + kt*512), gg);
      go = wmma_bf16(a0, load_b_swz(Wih1 + tO + kt*512), go);
      v16bf a1 = load_a_frag(&h1s[(m*16)*128 + kt*32], 128);
      gi = wmma_bf16(a1, load_b_swz(Whh1 + tI + kt*512), gi);
      gf = wmma_bf16(a1, load_b_swz(Whh1 + tF + kt*512), gf);
      gg = wmma_bf16(a1, load_b_swz(Whh1 + tG + kt*512), gg);
      go = wmma_bf16(a1, load_b_swz(Whh1 + tO + kt*512), go);
    }
    float h1v[8];
    #pragma unroll
    for (int e = 0; e < 8; ++e){
      float cc = sigm(gf[e])*c1[e] + sigm(gi[e])*tanhf(gg[e]);
      c1[e] = cc;
      h1v[e] = sigm(go[e])*tanhf(cc);
    }
    __syncthreads();                      // all h1_prev reads done
    #pragma unroll
    for (int e = 0; e < 8; ++e)
      h1s[(rloc0+e)*128 + u*16 + n] = (bf16)h1v[e];
    __syncthreads();

    // ---- output projection: out = h1 @ outW.T + outB, sigmoid on ch 0,1 ----
    if (threadIdx.x < 96){
      int r = threadIdx.x / 3, j = threadIdx.x % 3;
      float s = outB[j];
      #pragma unroll 8
      for (int k = 0; k < 128; ++k)
        s += (float)h1s[r*128 + k] * outW[j*128 + k];
      if (j < 2) s = sigm(s);
      out[((bbase + r)*NN + step)*3 + j] = s;
    }
    __syncthreads();
  }
}

// ---------------- host launch ----------------
extern "C" void kernel_launch(void* const* d_in, const int* in_sizes, int n_in,
                              void* d_out, int out_size, void* d_ws, size_t ws_size,
                              hipStream_t stream){
  (void)in_sizes; (void)n_in; (void)out_size; (void)ws_size;
  const float* note_input = (const float*)d_in[0];
  const float* beat_in    = (const float*)d_in[1];
  const float* cond       = (const float*)d_in[2];
  const float* beat_W     = (const float*)d_in[3];
  const float* beat_b     = (const float*)d_in[4];
  const float* vic_W      = (const float*)d_in[5];
  const float* vic_b      = (const float*)d_in[6];
  const float* ta_Wih0    = (const float*)d_in[7];
  // d_in[8] = ta_Whh0: dead (zero initial h)
  const float* ta_bih0    = (const float*)d_in[9];
  const float* ta_bhh0    = (const float*)d_in[10];
  const float* ta_Wih1    = (const float*)d_in[11];
  // d_in[12] = ta_Whh1: dead
  const float* ta_bih1    = (const float*)d_in[13];
  const float* ta_bhh1    = (const float*)d_in[14];
  const float* na_Wih0    = (const float*)d_in[15];
  const float* na_Whh0    = (const float*)d_in[16];
  const float* na_bih0    = (const float*)d_in[17];
  const float* na_bhh0    = (const float*)d_in[18];
  const float* na_Wih1    = (const float*)d_in[19];
  const float* na_Whh1    = (const float*)d_in[20];
  const float* na_bih1    = (const float*)d_in[21];
  const float* na_bhh1    = (const float*)d_in[22];
  const float* out_W      = (const float*)d_in[23];
  const float* out_b      = (const float*)d_in[24];

  char* ws = (char*)d_ws;
  size_t off = 0;
  auto alloc = [&](size_t bytes)->void*{
    void* p = ws + off;
    off = (off + bytes + 255) & ~(size_t)255;
    return p;
  };
  bf16*  feat     = (bf16*)alloc((size_t)ROWS*96*2);
  bf16*  w_ta0    = (bf16*)alloc((size_t)96*1024*2);
  bf16*  w_ta1    = (bf16*)alloc((size_t)256*1024*2);
  bf16*  w_nax    = (bf16*)alloc((size_t)256*512*2);
  bf16*  w_nah0   = (bf16*)alloc((size_t)128*512*2);
  bf16*  w_nai1   = (bf16*)alloc((size_t)128*512*2);
  bf16*  w_nah1   = (bf16*)alloc((size_t)128*512*2);
  float* tail     = (float*)alloc((size_t)3*512*4);
  float* bias_ta0 = (float*)alloc((size_t)1024*4);
  float* bias_ta1 = (float*)alloc((size_t)1024*4);
  float* bias_na0 = (float*)alloc((size_t)512*4);
  float* bias_na1 = (float*)alloc((size_t)512*4);
  bf16*  h0buf    = (bf16*)alloc((size_t)ROWS*256*2);
  bf16*  nf       = (bf16*)alloc((size_t)ROWS*256*2);
  float* xproj    = (float*)alloc((size_t)ROWS*512*4);

  auto g1 = [](long total){ return dim3((unsigned)((total + 255) / 256)); };

  // weight packing into WMMA-fragment-major layout (tiny)
  pack_w_swz<<<g1(96*1024),  256, 0, stream>>>(ta_Wih0, w_ta0, 1024,  73,  96);
  pack_w_swz<<<g1(256*1024), 256, 0, stream>>>(ta_Wih1, w_ta1, 1024, 256, 256);
  pack_w_swz<<<g1(256*512),  256, 0, stream>>>(na_Wih0, w_nax,  512, 259, 256); // first 256 K cols
  pack_w_swz<<<g1(128*512),  256, 0, stream>>>(na_Whh0, w_nah0, 512, 128, 128);
  pack_w_swz<<<g1(128*512),  256, 0, stream>>>(na_Wih1, w_nai1, 512, 128, 128);
  pack_w_swz<<<g1(128*512),  256, 0, stream>>>(na_Whh1, w_nah1, 512, 128, 128);
  combine_bias<<<g1(1024), 256, 0, stream>>>(ta_bih0, ta_bhh0, bias_ta0, 1024);
  combine_bias<<<g1(1024), 256, 0, stream>>>(ta_bih1, ta_bhh1, bias_ta1, 1024);
  combine_bias<<<g1(512),  256, 0, stream>>>(na_bih0, na_bhh0, bias_na0, 512);
  combine_bias<<<g1(512),  256, 0, stream>>>(na_bih1, na_bhh1, bias_na1, 512);
  pack_tail<<<g1(3*512),   256, 0, stream>>>(na_Wih0, tail);

  // feature assembly
  build_features<<<g1((long)ROWS*96), 256, 0, stream>>>(
      note_input, beat_in, beat_W, beat_b, vic_W, vic_b, feat);

  // TimeAxis: two zero-state LSTM layers as fused WMMA GEMMs
  lstm_zero_gemm<96> <<<dim3(ROWS/16), 512, 0, stream>>>(feat,  w_ta0, bias_ta0, h0buf);
  lstm_zero_gemm<256><<<dim3(ROWS/16), 512, 0, stream>>>(h0buf, w_ta1, bias_ta1, nf);

  // NoteAxis hoisted input projection (parallel GEMM)
  gemm_xproj<<<dim3(ROWS/16), 512, 0, stream>>>(nf, w_nax, xproj);

  // NoteAxis 48-step persistent recurrence + fused output projection
  note_axis_kernel<<<dim3(32), 512, 0, stream>>>(
      xproj, w_nah0, w_nai1, w_nah1, bias_na0, bias_na1, tail,
      cond, out_W, out_b, (float*)d_out);
}
